// KAN_Convolutional_Layer_46119358825214
// MI455X (gfx1250) — matrix operand order repaired and checked by reading, not compiled, not run on hardware
//
#include <hip/hip_runtime.h>

typedef _Float16 v16h __attribute__((ext_vector_type(16)));
typedef _Float16 h8   __attribute__((ext_vector_type(8)));
typedef float    v8f  __attribute__((ext_vector_type(8)));

constexpr int IN_H  = 96;
constexpr int IN_W  = 96;
constexpr int OUT_H = 94;
constexpr int OUT_W = 94;
constexpr int NCONV = 4;
constexpr int NB    = 8;          // grid_size + spline_order = 5 + 3
constexpr int TILE  = 32;         // output tile side
constexpr int FT    = TILE + 2;   // 34: feature-pixel tile side
constexpr int FSTR  = 36;         // padded feature row stride (pixels)
constexpr int NSLOT = 16;         // f16 slots per pixel: 8 bases, silu, 7 zeros

__global__ __launch_bounds__(256, 1)
void kan_conv_wmma_kernel(const float* __restrict__ x,
                          const float* __restrict__ base_w,
                          const float* __restrict__ spline_w,
                          const float* __restrict__ spline_scaler,
                          const float* __restrict__ grid,
                          float* __restrict__ out)
{
    __shared__ __align__(16) _Float16 feats[FT][FSTR][NSLOT];   // 38.25 KB
    __shared__ __align__(16) _Float16 wlds[16][16][NSLOT];      // 8 KB (tap>=9, c>=4 rows zero)

    const int plane = blockIdx.z;            // b*16 + c_in
    const int oy0   = blockIdx.y * TILE;
    const int ox0   = blockIdx.x * TILE;
    const int tid   = threadIdx.x;
    const int lane  = tid & 31;
    const int wave  = tid >> 5;
    const int cN    = lane & 15;             // WMMA column (conv idx if <4)
    const int hi    = lane >> 4;             // which K-half this lane owns

    // ---- grid knots (12 floats, uniform h = 0.4) ----
    float g[12];
#pragma unroll
    for (int j = 0; j < 12; ++j) g[j] = grid[j];

    // ---- stage 1: scaled weights -> LDS. One 16-slot row per thread, straight-line.
    // Row (c, tap): slots 0..7 = spline_w*scaler, slot 8 = base_w, 9..15 = 0.
    // Invalid rows (c>=4 or tap>=9) are zeroed through sc/bwv (finite products).
    {
        int  c   = tid >> 4;                  // 0..15
        int  tap = tid & 15;                  // 0..15
        bool vw  = (c < NCONV) && (tap < 9);
        int  wi  = vw ? (c * 9 + tap) : 0;    // clamped, always in-range
        float sc  = spline_scaler[wi];
        float bwv = base_w[wi];
        sc  = vw ? sc  : 0.0f;
        bwv = vw ? bwv : 0.0f;
        const float4* swp = (const float4*)(spline_w + wi * NB);   // 32B aligned
        float4 s0 = swp[0], s1 = swp[1];
        h8 lo, hh;
        lo[0] = (_Float16)(s0.x * sc); lo[1] = (_Float16)(s0.y * sc);
        lo[2] = (_Float16)(s0.z * sc); lo[3] = (_Float16)(s0.w * sc);
        lo[4] = (_Float16)(s1.x * sc); lo[5] = (_Float16)(s1.y * sc);
        lo[6] = (_Float16)(s1.z * sc); lo[7] = (_Float16)(s1.w * sc);
        hh[0] = (_Float16)bwv;
#pragma unroll
        for (int e = 1; e < 8; ++e) hh[e] = (_Float16)0.0f;
        *(h8*)&wlds[c][tap][0] = lo;    // ds_store_b128
        *(h8*)&wlds[c][tap][8] = hh;    // ds_store_b128
    }

    // ---- stage 2: per-pixel features (8 cubic B-spline bases + silu) -> LDS f16 ----
    const float* xp = x + (size_t)plane * IN_H * IN_W;
    for (int idx = tid; idx < FT * FT; idx += 256) {
        int fy = idx / FT, fx = idx - fy * FT;
        int gy = oy0 + fy, gx = ox0 + fx;
        bool inb = (gy < IN_H) && (gx < IN_W);
        int  cgy = inb ? gy : (IN_H - 1);
        int  cgx = inb ? gx : (IN_W - 1);
        float v  = xp[cgy * IN_W + cgx];
        v = inb ? v : 1.0e9f;           // outside knot range => every basis evaluates to 0
        float vs = inb ? v : 0.0f;      // silu input; silu(0) == 0 (keeps f16 finite)

        float t[11];
#pragma unroll
        for (int j = 0; j < 11; ++j)
            t[j] = (v >= g[j] && v < g[j + 1]) ? 1.0f : 0.0f;
#pragma unroll
        for (int p = 1; p <= 3; ++p) {
            float inv = 1.0f / (g[p] - g[0]);   // uniform knots: g[j+p]-g[j] = p*h
#pragma unroll
            for (int j = 0; j + p < 11; ++j)
                t[j] = (v - g[j]) * inv * t[j] + (g[j + p + 1] - v) * inv * t[j + 1];
        }
        float silu = vs / (1.0f + __expf(-vs));

        h8 lo, hh;
#pragma unroll
        for (int e = 0; e < 8; ++e) lo[e] = (_Float16)t[e];
        hh[0] = (_Float16)silu;
#pragma unroll
        for (int e = 1; e < 8; ++e) hh[e] = (_Float16)0.0f;
        *(h8*)&feats[fy][fx][0] = lo;   // ds_store_b128
        *(h8*)&feats[fy][fx][8] = hh;   // ds_store_b128
    }
    __syncthreads();

    // ---- B operands from LDS: step s covers taps (2s, 2s+1); tap 9 row = zeros ----
    // B 32x16 f16: lane<16 -> N=lane, K=0..15 ; lane>=16 -> N=lane-16, K=16..31
    v16h bmat[5];
#pragma unroll
    for (int s = 0; s < 5; ++s) {
        int tap = 2 * s + hi;                 // 0..9
        union { v16h v; h8 h[2]; } b;
        b.h[0] = *(const h8*)&wlds[cN][tap][0];
        b.h[1] = *(const h8*)&wlds[cN][tap][8];
        bmat[s] = b.v;
    }

    // ---- stage 3: WMMA contraction. 64 groups of 16 pixels; 8 per wave ----
    // A 16x32 f16: lane<16 -> M=lane, K=0..15 ; lane>=16 -> M=lane-16, K=16..31
    for (int gi = 0; gi < 8; ++gi) {
        int G   = wave * 8 + gi;
        int py  = G >> 1;                     // tile row 0..31
        int px0 = (G & 1) * 16;               // tile col base 0 or 16
        v8f acc = {};
#pragma unroll
        for (int s = 0; s < 5; ++s) {
            int tA = 2 * s;
            int tB = (s == 4) ? 8 : 2 * s + 1;   // step 4: B's upper-K columns are zero
            int dyA = tA / 3, dxA = tA % 3;
            int dyB = tB / 3, dxB = tB % 3;
            union { v16h v; h8 h[2]; } a;
            a.h[0] = *(const h8*)&feats[py + (hi ? dyB : dyA)][px0 + cN + (hi ? dxB : dxA)][0];
            a.h[1] = *(const h8*)&feats[py + (hi ? dyB : dyA)][px0 + cN + (hi ? dxB : dxA)][8];
            acc = __builtin_amdgcn_wmma_f32_16x16x32_f16(
                false, a.v, false, bmat[s], (short)0, acc, false, false);
        }
        // D 16x16 f32: lane = N + 16*(M/8), VGPR r = M%8  ->  M = hi*8 + r
        if (cN < NCONV) {
            int oy = oy0 + py;
            if (oy < OUT_H) {
                float* op  = out + (((size_t)plane * NCONV + cN) * OUT_H + oy) * OUT_W;
                int    oxb = ox0 + px0 + hi * 8;
                if (oxb + 8 <= OUT_W) {
                    // row base and oxb are both even -> 8-byte aligned b64 stores
                    float2* op2 = (float2*)(op + oxb);
#pragma unroll
                    for (int q = 0; q < 4; ++q)
                        op2[q] = make_float2(acc[2 * q], acc[2 * q + 1]);
                } else {
#pragma unroll
                    for (int r = 0; r < 8; ++r) {
                        int ox = oxb + r;
                        if (ox < OUT_W) op[ox] = acc[r];
                    }
                }
            }
        }
    }
}

extern "C" void kernel_launch(void* const* d_in, const int* in_sizes, int n_in,
                              void* d_out, int out_size, void* d_ws, size_t ws_size,
                              hipStream_t stream) {
    const float* x_in = (const float*)d_in[0];
    const float* bw   = (const float*)d_in[1];
    const float* sw   = (const float*)d_in[2];
    const float* ss   = (const float*)d_in[3];
    const float* gr   = (const float*)d_in[4];
    float* out        = (float*)d_out;

    dim3 grid((OUT_W + TILE - 1) / TILE,   // 3
              (OUT_H + TILE - 1) / TILE,   // 3
              8 * 16);                     // B*C planes
    kan_conv_wmma_kernel<<<grid, dim3(256), 0, stream>>>(x_in, bw, sw, ss, gr, out);
}
